// LowrankLearnableHash_58901181497379
// MI455X (gfx1250) — matrix operand order repaired and checked by reading, not compiled, not run on hardware
//
#include <hip/hip_runtime.h>
#include <hip/hip_bf16.h>

#define RESG   256
#define RADIUS 1.3f
#define NS     128
#define STEPSZ (2.0f * RADIUS / (float)NS)

typedef __attribute__((ext_vector_type(16))) _Float16 v16h;
typedef __attribute__((ext_vector_type(8)))  float    v8f;

__device__ __forceinline__ void bilin_setup(float cx, float cy, int W, int H,
                                            int& x0, int& y0, float& wx, float& wy) {
  float u = fminf(fmaxf((cx + 1.f) * 0.5f * (float)(W - 1), 0.f), (float)(W - 1));
  float v = fminf(fmaxf((cy + 1.f) * 0.5f * (float)(H - 1), 0.f), (float)(H - 1));
  float fx = fminf(floorf(u), (float)(W - 2));
  float fy = fminf(floorf(v), (float)(H - 2));
  x0 = (int)fx; y0 = (int)fy;
  wx = u - fx;  wy = v - fy;
}

__device__ __forceinline__ float bilin_fetch(const float* __restrict__ base, int W,
                                             float wx, float wy) {
  float g00 = base[0], g01 = base[1], g10 = base[W], g11 = base[W + 1];
  return g00 * (1.f - wx) * (1.f - wy) + g01 * wx * (1.f - wy)
       + g10 * (1.f - wx) * wy        + g11 * wx * wy;
}

// A fragment (16x32 f16): lane m=lane&15 is row M; hi selects K sub-block.
// K(e) = (e>=8 ? 16 : 0) + 8*hi + (e&7)   [CDNA5 ISA 7.12.2]
__device__ __forceinline__ v16h load_a_frag(const _Float16* act, int stride, int k0, int lane) {
  int m = lane & 15, hi = lane >> 4;
  const _Float16* p0 = act + m * stride + k0 + 8 * hi;
  const _Float16* p1 = p0 + 16;
  v16h a;
#pragma unroll
  for (int e = 0; e < 8; ++e) { a[e] = p0[e]; a[8 + e] = p1[e]; }
  return a;
}

// B fragment (32x16 f16): lane n=lane&15 is column N; K(e) same as A fragment.
__device__ __forceinline__ v16h load_b_frag(const _Float16* w, int ncols, int k0, int n0, int lane) {
  int n = lane & 15, hi = lane >> 4;
  const _Float16* p0 = w + (size_t)(k0 + 8 * hi) * ncols + n0 + n;
  const _Float16* p1 = p0 + (size_t)16 * ncols;
  v16h b;
#pragma unroll
  for (int e = 0; e < 8; ++e) { b[e] = p0[(size_t)e * ncols]; b[8 + e] = p1[(size_t)e * ncols]; }
  return b;
}

__device__ __forceinline__ v8f wmma_f16(v16h a, v16h b, v8f c) {
  return __builtin_amdgcn_wmma_f32_16x16x32_f16(false, a, false, b, (short)0, c, false, false);
}

__global__ void __launch_bounds__(256)
nerf_lowrank_fused(const float* __restrict__ rays_o, const float* __restrict__ rays_d,
                   const float* __restrict__ G,      const float* __restrict__ Fgrid,
                   const float* __restrict__ sigma_W1, const float* __restrict__ sigma_W2,
                   const float* __restrict__ color_W1, const float* __restrict__ color_W2,
                   const float* __restrict__ color_W3, float* __restrict__ out) {
  __shared__ _Float16 sW1s[32 * 64];   // sigma_W1  [K=32][N=64]
  __shared__ _Float16 sW2s[64 * 16];   // sigma_W2  [K=64][N=16]
  __shared__ _Float16 sW1c[32 * 64];   // color_W1  padded K 31->32
  __shared__ _Float16 sW2c[64 * 64];   // color_W2
  __shared__ _Float16 sW3c[64 * 16];   // color_W3  padded N 3->16
  __shared__ float sGv [NS * 2];       // plane-product coords
  __shared__ float sSig[NS];
  __shared__ float sMsk[NS];
  __shared__ float sCol[NS * 3];
  __shared__ _Float16 bufA[8][16 * 32];  // Fvals activations (per wave)
  __shared__ _Float16 bufB[8][16 * 64];  // hidden activations (per wave)
  __shared__ _Float16 bufC[8][16 * 32];  // [enc(16) | feat(15) | 0]

  const int t    = threadIdx.x;
  const int wv   = t >> 5;
  const int lane = t & 31;
  const int m    = lane & 15;
  const int hi   = lane >> 4;
  const int ray  = blockIdx.x;
  const int s    = wv * 16 + m;        // sample 0..127

  // ---- weights -> LDS (f16), with padding ----
  for (int i = t; i < 32 * 64; i += 256) sW1s[i] = (_Float16)sigma_W1[i];
  for (int i = t; i < 64 * 16; i += 256) sW2s[i] = (_Float16)sigma_W2[i];
  for (int i = t; i < 32 * 64; i += 256) {
    int r = i >> 6, c = i & 63;
    sW1c[i] = (_Float16)(r < 31 ? color_W1[r * 64 + c] : 0.f);
  }
  for (int i = t; i < 64 * 64; i += 256) sW2c[i] = (_Float16)color_W2[i];
  for (int i = t; i < 64 * 16; i += 256) {
    int r = i >> 4, c = i & 15;
    sW3c[i] = (_Float16)(c < 3 ? color_W3[r * 3 + c] : 0.f);
  }

  // ---- ray setup (cheap, redundant per thread; hits L0/L2) ----
  float ox = rays_o[ray * 3 + 0], oy = rays_o[ray * 3 + 1], oz = rays_o[ray * 3 + 2];
  float dx = rays_d[ray * 3 + 0], dy = rays_d[ray * 3 + 1], dz = rays_d[ray * 3 + 2];
  float inv = 1.f / sqrtf(dx * dx + dy * dy + dz * dz);
  dx *= inv; dy *= inv; dz *= inv;
  float bq   = ox * dx + oy * dy + oz * dz;
  float cq   = ox * ox + oy * oy + oz * oz - RADIUS * RADIUS;
  float disc = bq * bq - cq;
  float sq   = sqrtf(fmaxf(disc, 0.f));
  float tnear = fmaxf(-bq - sq, 0.f);
  float tfar  = -bq + sq;

  float tmid = tnear + ((float)s + 0.5f) * STEPSZ;
  int   msk  = (disc > 0.f) && (tmid <= tfar);
  if (hi == 0) sMsk[s] = msk ? 1.f : 0.f;

  float px = (ox + dx * tmid) / RADIUS;
  float py = (oy + dy * tmid) / RADIUS;
  float pz = (oz + dz * tmid) / RADIUS;

  __syncthreads();

  // ---- tri-plane sampling: lane handles dim=hi (ranks 0..7) for its point ----
  {
    float cxs[3] = {px, px, py};
    float cys[3] = {py, pz, pz};
    int x0[3], y0[3]; float wx[3], wy[3];
#pragma unroll
    for (int pl = 0; pl < 3; ++pl)
      bilin_setup(cxs[pl], cys[pl], RESG, RESG, x0[pl], y0[pl], wx[pl], wy[pl]);
    float gsum = 0.f;
#pragma unroll
    for (int r = 0; r < 8; ++r) {
      int ch = hi * 8 + r;                     // channel = dim*8 + rank
      float prod = 1.f;
#pragma unroll
      for (int pl = 0; pl < 3; ++pl) {
        const float* base = G + ((size_t)(pl * 16 + ch) * RESG + y0[pl]) * RESG + x0[pl];
        prod *= bilin_fetch(base, RESG, wx[pl], wy[pl]);
      }
      gsum += prod;
    }
    sGv[s * 2 + hi] = gsum;
  }
  __syncthreads();

  // ---- Fgrid sampling -> bufA (f16 16x32) ; SH deg-4 encoding -> bufC[:,0:16] ----
  {
    float gx = sGv[s * 2 + 0], gy = sGv[s * 2 + 1];
    int x0, y0; float wx, wy;
    bilin_setup(gx, gy, 64, 64, x0, y0, wx, wy);
#pragma unroll
    for (int j = 0; j < 16; ++j) {
      int ch = hi * 16 + j;
      const float* base = Fgrid + ((size_t)ch * 64 + y0) * 64 + x0;
      bufA[wv][m * 32 + ch] = (_Float16)bilin_fetch(base, 64, wx, wy);
    }
    if (hi == 0) {
      float x = dx, y = dy, z = dz;
      float xx = x * x, yy = y * y, zz = z * z;
      float xy = x * y, yz = y * z, xz = x * z;
      float enc[16];
      enc[0]  = 0.28209479177387814f;
      enc[1]  = -0.48860251190291987f * y;
      enc[2]  =  0.48860251190291987f * z;
      enc[3]  = -0.48860251190291987f * x;
      enc[4]  =  1.0925484305920792f * xy;
      enc[5]  = -1.0925484305920792f * yz;
      enc[6]  =  0.94617469575756f * zz - 0.31539156525252f;
      enc[7]  = -1.0925484305920792f * xz;
      enc[8]  =  0.5462742152960396f * (xx - yy);
      enc[9]  =  0.5900435899266435f * y * (-3.f * xx + yy);
      enc[10] =  2.890611442640554f * xy * z;
      enc[11] =  0.4570457994644657f * y * (1.f - 5.f * zz);
      enc[12] =  0.3731763325901154f * z * (5.f * zz - 3.f);
      enc[13] =  0.4570457994644657f * x * (1.f - 5.f * zz);
      enc[14] =  1.445305721320277f * z * (xx - yy);
      enc[15] =  0.5900435899266435f * x * (-xx + 3.f * yy);
#pragma unroll
      for (int j = 0; j < 16; ++j) bufC[wv][m * 32 + j] = (_Float16)enc[j];
    }
  }
  __syncthreads();

  // ---- sigma layer 1: [16x32]x[32x64] + relu -> bufB ----
  {
    v16h a = load_a_frag(bufA[wv], 32, 0, lane);
#pragma unroll
    for (int n = 0; n < 4; ++n) {
      v16h b = load_b_frag(sW1s, 64, 0, n * 16, lane);
      v8f c = {};
      c = wmma_f16(a, b, c);
      int nn = lane & 15, h2 = lane >> 4;
#pragma unroll
      for (int r = 0; r < 8; ++r)
        bufB[wv][(r + 8 * h2) * 64 + n * 16 + nn] = (_Float16)fmaxf(c[r], 0.f);
    }
  }
  __syncthreads();

  // ---- sigma layer 2: [16x64]x[64x16] -> sigma + feat (raw, no relu) ----
  {
    v16h a0 = load_a_frag(bufB[wv], 64, 0, lane);
    v16h a1 = load_a_frag(bufB[wv], 64, 32, lane);
    v8f c = {};
    c = wmma_f16(a0, load_b_frag(sW2s, 16, 0, 0, lane), c);
    c = wmma_f16(a1, load_b_frag(sW2s, 16, 32, 0, lane), c);
    int nn = lane & 15, h2 = lane >> 4;
    if (nn == 0) {
#pragma unroll
      for (int r = 0; r < 8; ++r) {
        sSig[wv * 16 + r + 8 * h2] = c[r];
        bufC[wv][(r + 8 * h2) * 32 + 31] = (_Float16)0.f;   // pad cin K=31
      }
    } else {
#pragma unroll
      for (int r = 0; r < 8; ++r)
        bufC[wv][(r + 8 * h2) * 32 + 16 + (nn - 1)] = (_Float16)c[r];
    }
  }
  __syncthreads();

  // ---- color layer 1: [16x32]x[32x64] + relu -> bufB ----
  {
    v16h a = load_a_frag(bufC[wv], 32, 0, lane);
#pragma unroll
    for (int n = 0; n < 4; ++n) {
      v16h b = load_b_frag(sW1c, 64, 0, n * 16, lane);
      v8f c = {};
      c = wmma_f16(a, b, c);
      int nn = lane & 15, h2 = lane >> 4;
#pragma unroll
      for (int r = 0; r < 8; ++r)
        bufB[wv][(r + 8 * h2) * 64 + n * 16 + nn] = (_Float16)fmaxf(c[r], 0.f);
    }
  }
  __syncthreads();

  // ---- color layer 2: [16x64]x[64x64] + relu, in-place via registers ----
  {
    v16h a0 = load_a_frag(bufB[wv], 64, 0, lane);
    v16h a1 = load_a_frag(bufB[wv], 64, 32, lane);
    v8f d[4];
#pragma unroll
    for (int n = 0; n < 4; ++n) {
      v8f c = {};
      c = wmma_f16(a0, load_b_frag(sW2c, 64, 0, n * 16, lane), c);
      c = wmma_f16(a1, load_b_frag(sW2c, 64, 32, n * 16, lane), c);
      d[n] = c;
    }
    __syncthreads();  // all A-frag reads complete before overwriting bufB
    int nn = lane & 15, h2 = lane >> 4;
#pragma unroll
    for (int n = 0; n < 4; ++n)
#pragma unroll
      for (int r = 0; r < 8; ++r)
        bufB[wv][(r + 8 * h2) * 64 + n * 16 + nn] = (_Float16)fmaxf(d[n][r], 0.f);
  }
  __syncthreads();

  // ---- color layer 3: [16x64]x[64x16] (cols 0..2 valid) ----
  {
    v16h a0 = load_a_frag(bufB[wv], 64, 0, lane);
    v16h a1 = load_a_frag(bufB[wv], 64, 32, lane);
    v8f c = {};
    c = wmma_f16(a0, load_b_frag(sW3c, 16, 0, 0, lane), c);
    c = wmma_f16(a1, load_b_frag(sW3c, 16, 32, 0, lane), c);
    int nn = lane & 15, h2 = lane >> 4;
    if (nn < 3) {
#pragma unroll
      for (int r = 0; r < 8; ++r)
        sCol[(wv * 16 + r + 8 * h2) * 3 + nn] = c[r];
    }
  }
  __syncthreads();

  // ---- per-ray transmittance scan + composite ----
  if (t == 0) {
    float T = 1.f, acc = 0.f, r0 = 0.f, r1 = 0.f, r2 = 0.f;
    for (int i = 0; i < NS; ++i) {
      float sg    = (sMsk[i] > 0.5f) ? fmaxf(sSig[i], 0.f) : 0.f;
      float alpha = 1.f - __expf(-sg * STEPSZ);
      float w     = alpha * T;
      r0  += w / (1.f + __expf(-sCol[i * 3 + 0]));
      r1  += w / (1.f + __expf(-sCol[i * 3 + 1]));
      r2  += w / (1.f + __expf(-sCol[i * 3 + 2]));
      acc += w;
      T   *= (1.f - alpha + 1e-10f);
    }
    out[ray * 3 + 0] = r0 + (1.f - acc);
    out[ray * 3 + 1] = r1 + (1.f - acc);
    out[ray * 3 + 2] = r2 + (1.f - acc);
  }
}

extern "C" void kernel_launch(void* const* d_in, const int* in_sizes, int n_in,
                              void* d_out, int out_size, void* d_ws, size_t ws_size,
                              hipStream_t stream) {
  const float* rays_o = (const float*)d_in[0];
  const float* rays_d = (const float*)d_in[1];
  const float* G      = (const float*)d_in[2];
  const float* Fgrid  = (const float*)d_in[3];
  const float* sW1    = (const float*)d_in[4];
  const float* sW2    = (const float*)d_in[5];
  const float* cW1    = (const float*)d_in[6];
  const float* cW2    = (const float*)d_in[7];
  const float* cW3    = (const float*)d_in[8];
  float* out = (float*)d_out;
  int B = in_sizes[0] / 3;   // 16384 rays
  hipLaunchKernelGGL(nerf_lowrank_fused, dim3(B), dim3(256), 0, stream,
                     rays_o, rays_d, G, Fgrid, sW1, sW2, cW1, cW2, cW3, out);
}
